// HetEncoder_18030272708835
// MI455X (gfx1250) — matrix operand order repaired and checked by reading, not compiled, not run on hardware
//
#include <hip/hip_runtime.h>
#include <cstdint>
#include <cstddef>

#define B_ROWS 4096
#define HID 512
#define EMB 128

typedef __bf16 bf16;
typedef __attribute__((ext_vector_type(16))) __bf16 v16bf;
typedef __attribute__((ext_vector_type(8)))  __bf16 v8bf;
typedef __attribute__((ext_vector_type(8)))  float  v8f;

union Frag16 { v16bf v; v8bf h[2]; };

__device__ __forceinline__ float wave_sum(float v) {
#pragma unroll
  for (int m = 16; m; m >>= 1) v += __shfl_xor(v, m, 32);
  return v;
}

// ---------------------------------------------------------------------------
// f32 -> bf16 straight conversion
__global__ __launch_bounds__(256) void cvt_bf16_kernel(
    const float* __restrict__ in, bf16* __restrict__ out, int n) {
  int i = blockIdx.x * 256 + threadIdx.x;
  if (i < n) out[i] = (bf16)in[i];
}

// W[K][N] f32 -> WT[N][K] bf16 (so WMMA B-fragments are contiguous loads)
__global__ __launch_bounds__(256) void transpose_bf16_kernel(
    const float* __restrict__ W, bf16* __restrict__ WT, int K, int N) {
  int idx = blockIdx.x * 256 + threadIdx.x;
  if (idx < K * N) {
    int k = idx / N, n = idx - k * N;
    WT[(size_t)n * K + k] = (bf16)W[idx];
  }
}

__global__ __launch_bounds__(256) void edge_type_kernel(
    const int* __restrict__ st, const int* __restrict__ dt,
    int* __restrict__ e, int n) {
  int i = blockIdx.x * 256 + threadIdx.x;
  if (i < n) e[i] = st[i] * 3 + dt[i];
}

// ---------------------------------------------------------------------------
// Direct (no-LDS) GEMM used for the small N=128 output projection.
// Block = 64 threads = 2 waves; each wave: 32(M) x 64(N). grid = (N/128, M/32).
__global__ __launch_bounds__(64) void gemm_bf16_wmma_kernel(
    const bf16* __restrict__ A, int lda,
    const bf16* __restrict__ WT,
    float* __restrict__ C, int ldc, int K)
{
  const int lane = threadIdx.x & 31;
  const int wave = threadIdx.x >> 5;
  const int m0 = blockIdx.y * 32;
  const int n0 = blockIdx.x * 128 + wave * 64;
  const int lhi = lane >> 4;
  const int llo = lane & 15;

  v8f acc[2][4];
#pragma unroll
  for (int mi = 0; mi < 2; ++mi)
#pragma unroll
    for (int ni = 0; ni < 4; ++ni)
#pragma unroll
      for (int r = 0; r < 8; ++r) acc[mi][ni][r] = 0.f;

  for (int k0 = 0; k0 < K; k0 += 32) {
    Frag16 aF[2], bF[4];
#pragma unroll
    for (int mi = 0; mi < 2; ++mi) {
      const bf16* pa = A + (size_t)(m0 + mi * 16 + llo) * lda + (k0 + lhi * 8);
      aF[mi].h[0] = *(const v8bf*)pa;
      aF[mi].h[1] = *(const v8bf*)(pa + 16);
    }
#pragma unroll
    for (int ni = 0; ni < 4; ++ni) {
      const bf16* pb = WT + (size_t)(n0 + ni * 16 + llo) * K + (k0 + lhi * 8);
      bF[ni].h[0] = *(const v8bf*)pb;
      bF[ni].h[1] = *(const v8bf*)(pb + 16);
    }
#pragma unroll
    for (int mi = 0; mi < 2; ++mi)
#pragma unroll
      for (int ni = 0; ni < 4; ++ni)
        acc[mi][ni] = __builtin_amdgcn_wmma_f32_16x16x32_bf16(
            false, aF[mi].v, false, bF[ni].v, (short)0, acc[mi][ni],
            false, false);
  }

#pragma unroll
  for (int mi = 0; mi < 2; ++mi)
#pragma unroll
    for (int ni = 0; ni < 4; ++ni) {
      int n = n0 + ni * 16 + llo;
      float* cp = C + (size_t)(m0 + mi * 16 + lhi * 8) * ldc + n;
#pragma unroll
      for (int r = 0; r < 8; ++r)
        cp[(size_t)r * ldc] = acc[mi][ni][r];
    }
}

// ---------------------------------------------------------------------------
// LDS-staged GEMM for N==512 (ldc==512). Block = 256 threads = 8 waves,
// tile = 64(M) x 256(N); B tile (256 cols x 32 K, bf16) double-buffered in LDS,
// filled with global_load_async_to_lds_b128 (ASYNCcnt), padded pitch for
// conflict-free ds_load_b128.
#define TILE_N 256
#define LDS_PITCH 40                 // bf16 elems per column row (80 bytes)
#define LDS_BUF_ELEMS (TILE_N * LDS_PITCH)

__global__ __launch_bounds__(256) void gemm_bf16_wmma_lds_kernel(
    const bf16* __restrict__ A, int lda,
    const bf16* __restrict__ WT,   // [512][K] bf16
    float* __restrict__ C, int K)
{
  __shared__ bf16 lds[2 * LDS_BUF_ELEMS];   // 40 KB

  const int tid  = threadIdx.x;
  const int lane = tid & 31;
  const int wave = tid >> 5;
  const int wm = wave >> 2;                 // 0..1 (M sub-block)
  const int wn = wave & 3;                  // 0..3 (N sub-block)
  const int m0  = blockIdx.y * 64 + wm * 32;
  const int n0g = blockIdx.x * TILE_N;
  const int lhi = lane >> 4;
  const int llo = lane & 15;

  // LDS byte offset of the staging buffer (flat->LDS aperture: low 32 bits).
  const unsigned lds_base = (unsigned)(uintptr_t)(&lds[0]);

  // Each thread stages one column (col = tid) of the 32-K B tile: 64 bytes
  // as 4 async 16-byte chunks.
  const bf16* gcol = WT + (size_t)(n0g + tid) * K;
  auto issue = [&](int buf, int k0) {
    unsigned l0 = lds_base + (unsigned)buf * (LDS_BUF_ELEMS * 2)
                + (unsigned)tid * (LDS_PITCH * 2);
    uint64_t g0 = (uint64_t)(uintptr_t)(gcol + k0);
#pragma unroll
    for (int kc = 0; kc < 4; ++kc) {
      unsigned la = l0 + kc * 16;
      uint64_t ga = g0 + kc * 16;
      asm volatile("global_load_async_to_lds_b128 %0, %1, off"
                   :: "v"(la), "v"(ga) : "memory");
    }
  };

  v8f acc[2][4];
#pragma unroll
  for (int mi = 0; mi < 2; ++mi)
#pragma unroll
    for (int ni = 0; ni < 4; ++ni)
#pragma unroll
      for (int r = 0; r < 8; ++r) acc[mi][ni][r] = 0.f;

  const int steps = K >> 5;
  issue(0, 0);

  for (int kt = 0; kt < steps; ++kt) {
    const int cur = kt & 1;
    const int k0 = kt << 5;
    const bool more = (kt + 1) < steps;
    if (more) issue(cur ^ 1, k0 + 32);

    if (more) asm volatile("s_wait_asynccnt 0x4" ::: "memory");
    else      asm volatile("s_wait_asynccnt 0x0" ::: "memory");
    __syncthreads();   // current tile visible to all waves

    Frag16 aF[2], bF[4];
#pragma unroll
    for (int mi = 0; mi < 2; ++mi) {
      const bf16* pa = A + (size_t)(m0 + mi * 16 + llo) * lda + (k0 + lhi * 8);
      aF[mi].h[0] = *(const v8bf*)pa;
      aF[mi].h[1] = *(const v8bf*)(pa + 16);
      if (more) __builtin_prefetch(pa + 32, 0, 0);
    }
#pragma unroll
    for (int ni = 0; ni < 4; ++ni) {
      const bf16* lp = &lds[(size_t)cur * LDS_BUF_ELEMS
                            + (size_t)(wn * 64 + ni * 16 + llo) * LDS_PITCH
                            + lhi * 8];
      bF[ni].h[0] = *(const v8bf*)lp;
      bF[ni].h[1] = *(const v8bf*)(lp + 16);
    }
#pragma unroll
    for (int mi = 0; mi < 2; ++mi)
#pragma unroll
      for (int ni = 0; ni < 4; ++ni)
        acc[mi][ni] = __builtin_amdgcn_wmma_f32_16x16x32_bf16(
            false, aF[mi].v, false, bF[ni].v, (short)0, acc[mi][ni],
            false, false);

    __syncthreads();   // all waves done reading before buffer is overwritten
  }

#pragma unroll
  for (int mi = 0; mi < 2; ++mi)
#pragma unroll
    for (int ni = 0; ni < 4; ++ni) {
      int n = n0g + wn * 64 + ni * 16 + llo;
      float* cp = C + (size_t)(m0 + mi * 16 + lhi * 8) * 512 + n;
#pragma unroll
      for (int r = 0; r < 8; ++r)
        cp[(size_t)r * 512] = acc[mi][ni][r];
    }
}

// ---------------------------------------------------------------------------
// y = [relu](LN(x + bias; g, b)); one wave per 512-wide row.
__global__ __launch_bounds__(256) void ln_act_kernel(
    const float* __restrict__ X, const float* __restrict__ bias,
    const float* __restrict__ g, const float* __restrict__ b,
    const int* __restrict__ rt, int sel, int do_relu,
    float* __restrict__ outF, bf16* __restrict__ outB)
{
  int row = blockIdx.x * 8 + (threadIdx.x >> 5);
  int lane = threadIdx.x & 31;
  if (sel >= 0 && rt[row] != sel) return;
  const float* xr = X + (size_t)row * HID;
  float x[16]; float s = 0.f;
#pragma unroll
  for (int j = 0; j < 16; ++j) { int c = lane + 32 * j; x[j] = xr[c] + bias[c]; s += x[j]; }
  float mean = wave_sum(s) * (1.f / HID);
  float vs = 0.f;
#pragma unroll
  for (int j = 0; j < 16; ++j) { float d = x[j] - mean; vs += d * d; }
  float rstd = rsqrtf(wave_sum(vs) * (1.f / HID) + 1e-5f);
#pragma unroll
  for (int j = 0; j < 16; ++j) {
    int c = lane + 32 * j;
    float y = (x[j] - mean) * rstd * g[c] + b[c];
    if (do_relu) y = fmaxf(y, 0.f);
    if (outF) outF[(size_t)row * HID + c] = y;
    if (outB) outB[(size_t)row * HID + c] = (bf16)y;
  }
}

// y = LN(dst + o; g, b) for rows with et[row]==e; bf16 output.
__global__ __launch_bounds__(256) void ln_res_kernel(
    const float* __restrict__ D, const float* __restrict__ O,
    const float* __restrict__ g, const float* __restrict__ b,
    const int* __restrict__ et, int e, bf16* __restrict__ outB)
{
  int row = blockIdx.x * 8 + (threadIdx.x >> 5);
  int lane = threadIdx.x & 31;
  if (et[row] != e) return;
  const float* dr = D + (size_t)row * HID;
  const float* orow = O + (size_t)row * HID;
  float x[16]; float s = 0.f;
#pragma unroll
  for (int j = 0; j < 16; ++j) { int c = lane + 32 * j; x[j] = dr[c] + orow[c]; s += x[j]; }
  float mean = wave_sum(s) * (1.f / HID);
  float vs = 0.f;
#pragma unroll
  for (int j = 0; j < 16; ++j) { float d = x[j] - mean; vs += d * d; }
  float rstd = rsqrtf(wave_sum(vs) * (1.f / HID) + 1e-5f);
#pragma unroll
  for (int j = 0; j < 16; ++j) {
    int c = lane + 32 * j;
    outB[(size_t)row * HID + c] = (bf16)((x[j] - mean) * rstd * g[c] + b[c]);
  }
}

// attn = sigmoid(dot(Q,K)/sqrt(128)) per head (4 heads of 128); msg = attn*V
__global__ __launch_bounds__(256) void attn_msg_kernel(
    const float* __restrict__ Q, const float* __restrict__ Km,
    const float* __restrict__ V, bf16* __restrict__ msg)
{
  int row = blockIdx.x * 8 + (threadIdx.x >> 5);
  int lane = threadIdx.x & 31;
  const size_t base = (size_t)row * HID;
  float vv[16]; float dot[4] = {0.f, 0.f, 0.f, 0.f};
#pragma unroll
  for (int j = 0; j < 16; ++j) {
    int c = lane + 32 * j;           // head = c/128 = j/4
    float q = Q[base + c], k = Km[base + c];
    vv[j] = V[base + c];
    dot[j >> 2] += q * k;
  }
  float att[4];
#pragma unroll
  for (int h = 0; h < 4; ++h) {
    float d = wave_sum(dot[h]);
    att[h] = 1.f / (1.f + __expf(-d * (1.f / 11.313708498984761f)));
  }
#pragma unroll
  for (int j = 0; j < 16; ++j) {
    int c = lane + 32 * j;
    msg[base + c] = (bf16)(att[j >> 2] * vv[j]);
  }
}

// z = LN(Z + ob; g, b); out = z / max(||z||2, 1e-12). One wave per 128-wide row.
__global__ __launch_bounds__(256) void emb_kernel(
    const float* __restrict__ Z, const float* __restrict__ ob,
    const float* __restrict__ g, const float* __restrict__ b,
    float* __restrict__ out)
{
  int row = blockIdx.x * 8 + (threadIdx.x >> 5);
  int lane = threadIdx.x & 31;
  const float* zr = Z + (size_t)row * EMB;
  float x[4]; float s = 0.f;
#pragma unroll
  for (int j = 0; j < 4; ++j) { int c = lane + 32 * j; x[j] = zr[c] + ob[c]; s += x[j]; }
  float mean = wave_sum(s) * (1.f / EMB);
  float vs = 0.f;
#pragma unroll
  for (int j = 0; j < 4; ++j) { float d = x[j] - mean; vs += d * d; }
  float rstd = rsqrtf(wave_sum(vs) * (1.f / EMB) + 1e-5f);
  float z[4]; float n2 = 0.f;
#pragma unroll
  for (int j = 0; j < 4; ++j) {
    int c = lane + 32 * j;
    z[j] = (x[j] - mean) * rstd * g[c] + b[c];
    n2 += z[j] * z[j];
  }
  float denom = fmaxf(sqrtf(wave_sum(n2)), 1e-12f);
#pragma unroll
  for (int j = 0; j < 4; ++j)
    out[(size_t)row * EMB + lane + 32 * j] = z[j] / denom;
}

// ---------------------------------------------------------------------------
extern "C" void kernel_launch(void* const* d_in, const int* in_sizes, int n_in,
                              void* d_out, int out_size, void* d_ws, size_t ws_size,
                              hipStream_t stream) {
  const float* anchor_feats = (const float*)d_in[0];
  const float* item_feats   = (const float*)d_in[1];
  const float* W1u  = (const float*)d_in[2];
  const float* W1e  = (const float*)d_in[3];
  const float* W1s  = (const float*)d_in[4];
  const float* b1   = (const float*)d_in[5];
  const float* ln1g = (const float*)d_in[6];
  const float* ln1b = (const float*)d_in[7];
  const float* W2   = (const float*)d_in[8];
  const float* b2   = (const float*)d_in[9];
  const float* ln2g = (const float*)d_in[10];
  const float* ln2b = (const float*)d_in[11];
  const float* Wk   = (const float*)d_in[12];
  const float* Wq   = (const float*)d_in[13];
  const float* Wv   = (const float*)d_in[14];
  const float* Wo   = (const float*)d_in[15];
  const float* hgtg = (const float*)d_in[16];
  const float* hgtb = (const float*)d_in[17];
  const float* outW = (const float*)d_in[18];
  const float* outb = (const float*)d_in[19];
  const float* olng = (const float*)d_in[20];
  const float* olnb = (const float*)d_in[21];
  const int*   a_ty = (const int*)d_in[22];
  const int*   i_ty = (const int*)d_in[23];
  float* out = (float*)d_out;

  const int B = B_ROWS;
  uint8_t* ws = (uint8_t*)d_ws;
  size_t off = 0;
  auto alloc = [&](size_t bytes) -> void* {
    void* p = ws + off;
    off += (bytes + 255) & ~(size_t)255;
    return p;
  };

  bf16* xaB = (bf16*)alloc((size_t)B * 256 * 2);
  bf16* xiB = (bf16*)alloc((size_t)B * 256 * 2);
  int K1[3] = {256, 192, 128};
  const float* W1p[3] = {W1u, W1e, W1s};
  bf16* WT1[3]; for (int t = 0; t < 3; ++t) WT1[t] = (bf16*)alloc((size_t)512 * K1[t] * 2);
  bf16* WT2[3]; for (int t = 0; t < 3; ++t) WT2[t] = (bf16*)alloc((size_t)512 * 512 * 2);
  bf16 *WTk[9], *WTq[9], *WTv[9];
  for (int e = 0; e < 9; ++e) WTk[e] = (bf16*)alloc((size_t)512 * 512 * 2);
  for (int e = 0; e < 9; ++e) WTq[e] = (bf16*)alloc((size_t)512 * 512 * 2);
  for (int e = 0; e < 9; ++e) WTv[e] = (bf16*)alloc((size_t)512 * 512 * 2);
  bf16* WTo[3]; for (int t = 0; t < 3; ++t) WTo[t] = (bf16*)alloc((size_t)512 * 512 * 2);
  bf16* WTe = (bf16*)alloc((size_t)128 * 512 * 2);

  float* tmpF = (float*)alloc((size_t)B * HID * 4);
  bf16*  tmpB = (bf16*)alloc((size_t)B * HID * 2);
  float* haF  = (float*)alloc((size_t)B * HID * 4);
  float* hiF  = (float*)alloc((size_t)B * HID * 4);
  bf16*  haB  = (bf16*)alloc((size_t)B * HID * 2);
  bf16*  hiB  = (bf16*)alloc((size_t)B * HID * 2);
  float* Kf   = (float*)alloc((size_t)B * HID * 4);
  float* Qf   = (float*)alloc((size_t)B * HID * 4);
  float* Vf   = (float*)alloc((size_t)B * HID * 4);
  float* Of   = (float*)alloc((size_t)B * HID * 4);
  bf16*  msgB = (bf16*)alloc((size_t)B * HID * 2);
  bf16*  hanB = (bf16*)alloc((size_t)B * HID * 2);
  bf16*  hinB = (bf16*)alloc((size_t)B * HID * 2);
  float* Zf   = (float*)alloc((size_t)B * EMB * 4);
  int*   eA   = (int*)alloc((size_t)B * 4);
  int*   eI   = (int*)alloc((size_t)B * 4);

  auto cvt = [&](const float* src, bf16* dst, int cnt) {
    cvt_bf16_kernel<<<dim3((cnt + 255) / 256), dim3(256), 0, stream>>>(src, dst, cnt);
  };
  auto tr = [&](const float* src, bf16* dst, int K, int N) {
    transpose_bf16_kernel<<<dim3((K * N + 255) / 256), dim3(256), 0, stream>>>(src, dst, K, N);
  };
  auto gemm = [&](const bf16* Ab, int lda, const bf16* Wt, int K, int N, float* C) {
    if (N == 512)
      gemm_bf16_wmma_lds_kernel<<<dim3(512 / TILE_N, B / 64), dim3(256), 0, stream>>>(
          Ab, lda, Wt, C, K);
    else
      gemm_bf16_wmma_kernel<<<dim3(N / 128, B / 32), dim3(64), 0, stream>>>(
          Ab, lda, Wt, C, N, K);
  };

  // ---- one-time conversions (re-run per call; deterministic) ----
  cvt(anchor_feats, xaB, B * 256);
  cvt(item_feats,   xiB, B * 256);
  for (int t = 0; t < 3; ++t) tr(W1p[t], WT1[t], K1[t], 512);
  for (int t = 0; t < 3; ++t) tr(W2 + (size_t)t * 512 * 512, WT2[t], 512, 512);
  for (int e = 0; e < 9; ++e) tr(Wk + (size_t)e * 512 * 512, WTk[e], 512, 512);
  for (int e = 0; e < 9; ++e) tr(Wq + (size_t)e * 512 * 512, WTq[e], 512, 512);
  for (int e = 0; e < 9; ++e) tr(Wv + (size_t)e * 512 * 512, WTv[e], 512, 512);
  for (int t = 0; t < 3; ++t) tr(Wo + (size_t)t * 512 * 512, WTo[t], 512, 512);
  tr(outW, WTe, 512, 128);

  edge_type_kernel<<<dim3(B / 256), dim3(256), 0, stream>>>(i_ty, a_ty, eA, B);
  edge_type_kernel<<<dim3(B / 256), dim3(256), 0, stream>>>(a_ty, i_ty, eI, B);

  // ---- encoders: all 3 type-branches over full batch, select per row ----
  auto encode = [&](const bf16* xB, const int* ty, float* hF, bf16* hB) {
    for (int t = 0; t < 3; ++t) {
      gemm(xB, 256, WT1[t], K1[t], 512, tmpF);
      ln_act_kernel<<<dim3(B / 8), dim3(256), 0, stream>>>(
          tmpF, b1 + t * 512, ln1g + t * 512, ln1b + t * 512,
          ty, -1, 1, (float*)nullptr, tmpB);
      gemm(tmpB, 512, WT2[t], 512, 512, tmpF);
      ln_act_kernel<<<dim3(B / 8), dim3(256), 0, stream>>>(
          tmpF, b2 + t * 512, ln2g + t * 512, ln2b + t * 512,
          ty, t, 1, hF, hB);
    }
  };
  encode(xaB, a_ty, haF, haB);
  encode(xiB, i_ty, hiF, hiB);

  // ---- HGT: all 9 edge branches, select per row by edge type ----
  auto hgt = [&](const bf16* srcB, const bf16* dstB, const float* dstF,
                 const int* eArr, bf16* outB) {
    for (int e = 0; e < 9; ++e) {
      gemm(srcB, 512, WTk[e], 512, 512, Kf);
      gemm(dstB, 512, WTq[e], 512, 512, Qf);
      gemm(srcB, 512, WTv[e], 512, 512, Vf);
      attn_msg_kernel<<<dim3(B / 8), dim3(256), 0, stream>>>(Qf, Kf, Vf, msgB);
      gemm(msgB, 512, WTo[e % 3], 512, 512, Of);
      ln_res_kernel<<<dim3(B / 8), dim3(256), 0, stream>>>(
          dstF, Of, hgtg + (e % 3) * 512, hgtb + (e % 3) * 512, eArr, e, outB);
    }
  };
  hgt(hiB, haB, haF, eA, hanB);  // item -> anchor updates h_a
  hgt(haB, hiB, hiF, eI, hinB);  // anchor -> item updates h_i

  // ---- output embeddings ----
  gemm(hanB, 512, WTe, 512, 128, Zf);
  emb_kernel<<<dim3(B / 8), dim3(256), 0, stream>>>(Zf, outb, olng, olnb, out);
  gemm(hinB, 512, WTe, 512, 128, Zf);
  emb_kernel<<<dim3(B / 8), dim3(256), 0, stream>>>(Zf, outb, olng, olnb,
                                                    out + (size_t)B * EMB);
}
